// ProteinMPNN_34351148434194
// MI455X (gfx1250) — matrix operand order repaired and checked by reading, not compiled, not run on hardware
//
#include <hip/hip_runtime.h>

#define B_  4
#define L_  2048
#define K_  48
#define H_  128
#define IN1 384      // H + NIN
#define FF_ 512
#define SCALE_ 30.0f
#define EPS_ 1e-5f

// padded LDS strides (elements): stride_dwords % 64 == 4 -> bank-conflict-free A loads,
// and multiples of 8 elements -> 16B alignment for v8bf/b128 access preserved.
#define AST  392   // A-tile row stride (bf16), data cols 0..383
#define TST  136   // act-tile row stride (bf16), data cols 0..127
#define FST  520   // FFN act-tile row stride (bf16), data cols 0..511
#define H1ST 132   // f32 residual tile row stride (floats)

// per-wave dynamic LDS: aIn 48*392*2 + tmp 48*136*2 + stg 16*128*4
#define AIN_B  37632
#define TMP_B  13056
#define STG_B  8192
#define WBYTES (AIN_B + TMP_B + STG_B)   // 58880

typedef __attribute__((ext_vector_type(16))) __bf16 v16bf;
typedef __attribute__((ext_vector_type(8)))  __bf16 v8bf;
typedef __attribute__((ext_vector_type(8)))  float  v8f;
typedef __attribute__((ext_vector_type(4)))  float  v4f;

// ---------- scalar helpers ----------
__device__ inline unsigned short f2bfu(float f) {
    union { float f; unsigned u; } x; x.f = f;
    unsigned u = x.u + 0x7FFFu + ((x.u >> 16) & 1u);   // round-to-nearest-even
    return (unsigned short)(u >> 16);
}
__device__ inline void store4bf(unsigned short* dst, float a, float b, float c, float d) {
    union { unsigned short us[4]; unsigned long long u; } p;
    p.us[0] = f2bfu(a); p.us[1] = f2bfu(b); p.us[2] = f2bfu(c); p.us[3] = f2bfu(d);
    *reinterpret_cast<unsigned long long*>(dst) = p.u;
}
__device__ inline float geluf(float x) {
    return 0.5f * x * (1.0f + erff(x * 0.70710678118654752f));
}
__device__ inline float grp16_sum(float v) {
    v += __shfl_xor(v, 1, 32);
    v += __shfl_xor(v, 2, 32);
    v += __shfl_xor(v, 4, 32);
    v += __shfl_xor(v, 8, 32);
    return v;
}
__device__ inline float wave_sum(float v) {
    v = grp16_sum(v);
    v += __shfl_xor(v, 16, 32);
    return v;
}

// ---------- CDNA5 async global->LDS (ASYNCcnt-tracked) ----------
__device__ inline void async_g2l_b128(unsigned ldsOff, const void* gaddr) {
    asm volatile("global_load_async_to_lds_b128 %0, %1, off"
                 :: "v"(ldsOff), "v"((unsigned long long)(size_t)gaddr)
                 : "memory");
}
__device__ inline void wait_async0() {
    asm volatile("s_wait_asynccnt 0x0" ::: "memory");
}
__device__ inline void wait_ds0() {
    asm volatile("s_wait_dscnt 0x0" ::: "memory");
}

// ---------- WMMA fragment helpers ----------
// 16-bit A 16x32 layout: lane<16 -> M=lane, halves0-7:K=kb+0..7, halves8-15:K=kb+16..23
//                        lane>=16 -> M=lane-16, K=kb+8..15 and kb+24..31
__device__ inline v16bf load_a_lds(const unsigned short* base, int lane, int stride, int kbase) {
    int m = lane & 15;
    int g = (lane >> 4) & 1;
    const unsigned short* p = base + m * stride + kbase + g * 8;
    v8bf lo = *reinterpret_cast<const v8bf*>(p);
    v8bf hi = *reinterpret_cast<const v8bf*>(p + 16);
    v16bf a;
#pragma unroll
    for (int i = 0; i < 8; ++i) { a[i] = lo[i]; a[8 + i] = hi[i]; }
    return a;
}
__device__ inline v16bf load_b_pk(const unsigned short* p) {
    v8bf lo = *reinterpret_cast<const v8bf*>(p);
    v8bf hi = *reinterpret_cast<const v8bf*>(p + 8);
    v16bf b;
#pragma unroll
    for (int i = 0; i < 8; ++i) { b[i] = lo[i]; b[8 + i] = hi[i]; }
    return b;
}
__device__ inline v8f wmma_bf16(v16bf a, v16bf b, v8f c) {
    return __builtin_amdgcn_wmma_f32_16x16x32_bf16(false, a, false, b, (short)0, c, false, false);
}

// GEMM: A[48 x ks*32] (3 M-tiles, bf16 LDS) x packed-W -> acc[3][8]; B fragment reused 3x
__device__ inline void gemm8x3(const unsigned short* aLds, int strideA, int ks,
                               const unsigned short* wPk, int ntt, int t0,
                               int lane, v8f acc[3][8]) {
#pragma unroll
    for (int ti = 0; ti < 3; ++ti)
#pragma unroll
        for (int j = 0; j < 8; ++j) acc[ti][j] = (v8f)0.0f;
    for (int s = 0; s < ks; ++s) {
        v16bf a0 = load_a_lds(aLds,                lane, strideA, s * 32);
        v16bf a1 = load_a_lds(aLds + 16 * strideA, lane, strideA, s * 32);
        v16bf a2 = load_a_lds(aLds + 32 * strideA, lane, strideA, s * 32);
#pragma unroll
        for (int j = 0; j < 8; ++j) {
            v16bf b = load_b_pk(wPk + ((size_t)(s * ntt + t0 + j)) * 512 + lane * 16);
            acc[0][j] = wmma_bf16(a0, b, acc[0][j]);
            acc[1][j] = wmma_bf16(a1, b, acc[1][j]);
            acc[2][j] = wmma_bf16(a2, b, acc[2][j]);
        }
    }
}
// 1-M-tile variant (FFN kernel)
__device__ inline void gemm8(const unsigned short* aLds, int strideA, int ks,
                             const unsigned short* wPk, int ntt, int t0,
                             int lane, v8f acc[8]) {
#pragma unroll
    for (int j = 0; j < 8; ++j) acc[j] = (v8f)0.0f;
    for (int s = 0; s < ks; ++s) {
        v16bf a = load_a_lds(aLds, lane, strideA, s * 32);
#pragma unroll
        for (int j = 0; j < 8; ++j) {
            const unsigned short* bp = wPk + ((size_t)(s * ntt + t0 + j)) * 512 + lane * 16;
            acc[j] = wmma_bf16(a, load_b_pk(bp), acc[j]);
        }
    }
}

// bias + gelu + store C tiles to LDS bf16 (3 M-tiles)
__device__ inline void act_store3(const v8f acc[3][8], const float* bias,
                                  unsigned short* tmp, int stride, int lane) {
    int g = (lane >> 4) & 1, nl = lane & 15;
#pragma unroll
    for (int j = 0; j < 8; ++j) {
        float bv = bias[j * 16 + nl];
#pragma unroll
        for (int ti = 0; ti < 3; ++ti)
#pragma unroll
            for (int r = 0; r < 8; ++r) {
                float v = geluf(acc[ti][j][r] + bv);
                tmp[(ti * 16 + r + 8 * g) * stride + j * 16 + nl] = f2bfu(v);
            }
    }
    wait_ds0();
}
__device__ inline void act_store(const v8f acc[8], const float* bias, int nbase,
                                 unsigned short* tmp, int stride, int lane) {
    int g = (lane >> 4) & 1, nl = lane & 15;
#pragma unroll
    for (int j = 0; j < 8; ++j) {
        float bv = bias[nbase + j * 16 + nl];
#pragma unroll
        for (int r = 0; r < 8; ++r) {
            float v = geluf(acc[j][r] + bv);
            tmp[(r + 8 * g) * stride + nbase + j * 16 + nl] = f2bfu(v);
        }
    }
    wait_ds0();
}

// build 48 x 384 bf16 input tile [hVrow | h_E | hV[nbr]] using async staging for h_E
__device__ inline void build_input(const float* __restrict__ hVsrc, const float* __restrict__ hE,
                                   const int* __restrict__ eIdx, size_t rowBL, size_t bBase,
                                   unsigned short* aIn, float* stg, unsigned stgOff, int lane) {
    // cols 0..127: broadcast h_V[b,l]
    v4f hv = reinterpret_cast<const v4f*>(hVsrc + rowBL * H_)[lane];
#pragma unroll 1
    for (int m = 0; m < 48; ++m)
        store4bf(aIn + m * AST + lane * 4, hv[0], hv[1], hv[2], hv[3]);
    // cols 128..255: h_E slab, async-staged in 3 x 8KB chunks
#pragma unroll 1
    for (int ch = 0; ch < 3; ++ch) {
        const float* src = hE + (rowBL * K_ + ch * 16) * (size_t)H_;  // 2048 contiguous f32
#pragma unroll 1
        for (int i = lane; i < 512; i += 32)
            async_g2l_b128(stgOff + i * 16, src + i * 4);
        wait_async0();
#pragma unroll 1
        for (int q = lane; q < 512; q += 32) {
            v4f v = reinterpret_cast<const v4f*>(stg)[q];
            int e = q * 4;
            int m = e >> 7, c = e & 127;
            store4bf(aIn + (ch * 16 + m) * AST + 128 + c, v[0], v[1], v[2], v[3]);
        }
        wait_ds0();   // protect stg reuse next chunk
    }
    // cols 256..383: gathered neighbor rows
#pragma unroll 1
    for (int m = 0; m < 48; ++m) {
        int nb = eIdx[rowBL * K_ + m];
        v4f vn = reinterpret_cast<const v4f*>(hVsrc + (bBase + nb) * (size_t)H_)[lane];
        store4bf(aIn + m * AST + 256 + lane * 4, vn[0], vn[1], vn[2], vn[3]);
    }
    wait_ds0();
}

// ---------- weight packing (f32 [O,I] torch layout -> bf16 B-fragment order) ----------
__global__ void k_pack(const float* __restrict__ w, unsigned short* __restrict__ dst, int O, int I) {
    int nt = O / 16, ks = I / 32;
    int total = ks * nt * 512;
    for (int p = blockIdx.x * blockDim.x + threadIdx.x; p < total; p += gridDim.x * blockDim.x) {
        int h    = p & 15;
        int lane = (p >> 4) & 31;
        int t    = (p >> 9) % nt;
        int s    = p / (512 * nt);
        int kk   = (h & 7) + 8 * (lane >> 4) + 16 * (h >> 3);   // B layout: K = (h&7) + 8*(lane>=16) + 16*(h>=8)
        int nl   = lane & 15;              //           N = lane&15
        dst[p] = f2bfu(w[(size_t)(t * 16 + nl) * I + (s * 32 + kk)]);
    }
}

// ---------- kernel 1: node message MLP (one wave = one (b,l), 48 tokens) ----------
__global__ __launch_bounds__(64) void k_node_msg(
    const float* __restrict__ hV, const float* __restrict__ hE, const int* __restrict__ eIdx,
    const float* __restrict__ maskAtt,
    const unsigned short* __restrict__ pW1, const float* __restrict__ b1,
    const unsigned short* __restrict__ pW2, const float* __restrict__ b2,
    const unsigned short* __restrict__ pW3, const float* __restrict__ b3,
    float* __restrict__ dh)
{
    extern __shared__ char smem[];
    int wv = threadIdx.x >> 5, lane = threadIdx.x & 31;
    char* wbase = smem + wv * WBYTES;
    unsigned short* aIn = (unsigned short*)wbase;
    unsigned short* tmp = (unsigned short*)(wbase + AIN_B);
    float*          stg = (float*)(wbase + AIN_B + TMP_B);
    unsigned stgOff = (unsigned)(size_t)(const void*)stg;

    size_t rowBL = (size_t)blockIdx.x * 2 + wv;                // 0 .. B*L-1
    size_t bBase = (rowBL / L_) * L_;

    build_input(hV, hE, eIdx, rowBL, bBase, aIn, stg, stgOff, lane);

    v8f acc[3][8];
    gemm8x3(aIn, AST, 12, pW1, 8, 0, lane, acc);
    act_store3(acc, b1, tmp, TST, lane);
    gemm8x3(tmp, TST, 4, pW2, 8, 0, lane, acc);
    act_store3(acc, b2, tmp, TST, lane);
    gemm8x3(tmp, TST, 4, pW3, 8, 0, lane, acc);

    // bias + mask_attend + full sum over 48 tokens -> plain store to dh
    int g = (lane >> 4) & 1, nl = lane & 15;
    float mk[3][8];
#pragma unroll
    for (int ti = 0; ti < 3; ++ti)
#pragma unroll
        for (int r = 0; r < 8; ++r)
            mk[ti][r] = maskAtt[rowBL * K_ + ti * 16 + r + 8 * g];
#pragma unroll
    for (int j = 0; j < 8; ++j) {
        float bv = b3[j * 16 + nl];
        float s = 0.f;
#pragma unroll
        for (int ti = 0; ti < 3; ++ti)
#pragma unroll
            for (int r = 0; r < 8; ++r) s += (acc[ti][j][r] + bv) * mk[ti][r];
        s += __shfl_xor(s, 16, 32);
        if (lane < 16) dh[rowBL * H_ + j * 16 + lane] = s;
    }
}

// ---------- kernel 2: residual+LN1, FFN (WMMA), residual+LN2, mask ----------
__global__ __launch_bounds__(64) void k_node_ffn(
    const float* __restrict__ hV, const float* __restrict__ dh, const float* __restrict__ maskV,
    const unsigned short* __restrict__ pWin, const float* __restrict__ bin,
    const unsigned short* __restrict__ pWout, const float* __restrict__ bout,
    const float* __restrict__ ln1g, const float* __restrict__ ln1b,
    const float* __restrict__ ln2g, const float* __restrict__ ln2b,
    float* __restrict__ hVout)
{
    __shared__ __align__(16) float          shH1[2][16 * H1ST];
    __shared__ __align__(16) unsigned short shA [2][16 * TST];
    __shared__ __align__(16) unsigned short shT [2][16 * FST];
    int wv = threadIdx.x >> 5, lane = threadIdx.x & 31;
    int tile = blockIdx.x * 2 + wv;
    size_t row0 = (size_t)tile * 16;
    float* h1 = shH1[wv];
    unsigned short* a16 = shA[wv];
    unsigned short* t16 = shT[wv];

    for (int m = 0; m < 16; ++m) {
        size_t row = row0 + m;
        float v[4]; float s = 0.f;
#pragma unroll
        for (int i = 0; i < 4; ++i) {
            int c = lane * 4 + i;
            v[i] = hV[row * H_ + c] + dh[row * H_ + c] * (1.0f / SCALE_);
            s += v[i];
        }
        float mean = wave_sum(s) * (1.0f / H_);
        float q = 0.f;
#pragma unroll
        for (int i = 0; i < 4; ++i) { float d = v[i] - mean; q += d * d; }
        float rstd = rsqrtf(wave_sum(q) * (1.0f / H_) + EPS_);
#pragma unroll
        for (int i = 0; i < 4; ++i) {
            int c = lane * 4 + i;
            float o = (v[i] - mean) * rstd * ln1g[c] + ln1b[c];
            h1[m * H1ST + c] = o;
            a16[m * TST + c] = f2bfu(o);
        }
    }
    wait_ds0();

    v8f acc[8];
    for (int ch = 0; ch < 4; ++ch) {
        gemm8(a16, TST, 4, pWin, 32, ch * 8, lane, acc);
        act_store(acc, bin, ch * 128, t16, FST, lane);
    }
    gemm8(t16, FST, 16, pWout, 8, 0, lane, acc);

    int g = (lane >> 4) & 1, nl = lane & 15;
    float rowm[8], rowr[8];
#pragma unroll
    for (int r = 0; r < 8; ++r) {
        int m = r + 8 * g;
        float s = 0.f;
#pragma unroll
        for (int j = 0; j < 8; ++j) {
            float t = acc[j][r] + bout[j * 16 + nl] + h1[m * H1ST + j * 16 + nl];
            acc[j][r] = t;
            s += t;
        }
        float mean = grp16_sum(s) * (1.0f / H_);
        float q = 0.f;
#pragma unroll
        for (int j = 0; j < 8; ++j) { float d = acc[j][r] - mean; q += d * d; }
        rowm[r] = mean;
        rowr[r] = rsqrtf(grp16_sum(q) * (1.0f / H_) + EPS_);
    }
#pragma unroll
    for (int r = 0; r < 8; ++r) {
        int m = r + 8 * g;
        size_t row = row0 + m;
        float mv = maskV[row];
#pragma unroll
        for (int j = 0; j < 8; ++j) {
            int n = j * 16 + nl;
            float o = ((acc[j][r] - rowm[r]) * rowr[r] * ln2g[n] + ln2b[n]) * mv;
            hVout[row * H_ + n] = o;
        }
    }
}

// ---------- kernel 3: edge message MLP + residual + LN3 (one wave = one (b,l)) ----------
__global__ __launch_bounds__(64) void k_edge(
    const float* __restrict__ hVn, const float* __restrict__ hE, const int* __restrict__ eIdx,
    const unsigned short* __restrict__ pW11, const float* __restrict__ b11,
    const unsigned short* __restrict__ pW12, const float* __restrict__ b12,
    const unsigned short* __restrict__ pW13, const float* __restrict__ b13,
    const float* __restrict__ ln3g, const float* __restrict__ ln3b,
    float* __restrict__ hEout)
{
    extern __shared__ char smem[];
    int wv = threadIdx.x >> 5, lane = threadIdx.x & 31;
    char* wbase = smem + wv * WBYTES;
    unsigned short* aIn = (unsigned short*)wbase;
    unsigned short* tmp = (unsigned short*)(wbase + AIN_B);
    float*          stg = (float*)(wbase + AIN_B + TMP_B);
    unsigned stgOff = (unsigned)(size_t)(const void*)stg;

    size_t rowBL = (size_t)blockIdx.x * 2 + wv;
    size_t bBase = (rowBL / L_) * L_;

    build_input(hVn, hE, eIdx, rowBL, bBase, aIn, stg, stgOff, lane);

    v8f acc[3][8];
    gemm8x3(aIn, AST, 12, pW11, 8, 0, lane, acc);
    act_store3(acc, b11, tmp, TST, lane);
    gemm8x3(tmp, TST, 4, pW12, 8, 0, lane, acc);
    act_store3(acc, b12, tmp, TST, lane);
    gemm8x3(tmp, TST, 4, pW13, 8, 0, lane, acc);

    // per-tile: residual + LN3 + store (keeps register pressure bounded)
    int g = (lane >> 4) & 1, nl = lane & 15;
#pragma unroll
    for (int ti = 0; ti < 3; ++ti) {
        float rowm[8], rowr[8];
#pragma unroll
        for (int r = 0; r < 8; ++r) {
            int m = ti * 16 + r + 8 * g;
            size_t erow = rowBL * K_ + m;
            float s = 0.f;
#pragma unroll
            for (int j = 0; j < 8; ++j) {
                float t = acc[ti][j][r] + b13[j * 16 + nl] + hE[erow * H_ + j * 16 + nl];
                acc[ti][j][r] = t;
                s += t;
            }
            float mean = grp16_sum(s) * (1.0f / H_);
            float q = 0.f;
#pragma unroll
            for (int j = 0; j < 8; ++j) { float d = acc[ti][j][r] - mean; q += d * d; }
            rowm[r] = mean;
            rowr[r] = rsqrtf(grp16_sum(q) * (1.0f / H_) + EPS_);
        }
#pragma unroll
        for (int r = 0; r < 8; ++r) {
            int m = ti * 16 + r + 8 * g;
            size_t erow = rowBL * K_ + m;
#pragma unroll
            for (int j = 0; j < 8; ++j) {
                int n = j * 16 + nl;
                hEout[erow * H_ + n] = (acc[ti][j][r] - rowm[r]) * rowr[r] * ln3g[n] + ln3b[n];
            }
        }
    }
}

// ---------- launcher ----------
extern "C" void kernel_launch(void* const* d_in, const int* in_sizes, int n_in,
                              void* d_out, int out_size, void* d_ws, size_t ws_size,
                              hipStream_t stream) {
    const float* hV      = (const float*)d_in[0];
    const float* hE      = (const float*)d_in[1];
    const int*   eIdx    = (const int*)  d_in[2];
    const float* maskV   = (const float*)d_in[3];
    const float* maskAtt = (const float*)d_in[4];
    const float* W1w  = (const float*)d_in[5],  *W1b  = (const float*)d_in[6];
    const float* W2w  = (const float*)d_in[7],  *W2b  = (const float*)d_in[8];
    const float* W3w  = (const float*)d_in[9],  *W3b  = (const float*)d_in[10];
    const float* W11w = (const float*)d_in[11], *W11b = (const float*)d_in[12];
    const float* W12w = (const float*)d_in[13], *W12b = (const float*)d_in[14];
    const float* W13w = (const float*)d_in[15], *W13b = (const float*)d_in[16];
    const float* Winw = (const float*)d_in[17], *Winb = (const float*)d_in[18];
    const float* Woutw= (const float*)d_in[19], *Woutb= (const float*)d_in[20];
    const float* ln1g = (const float*)d_in[21], *ln1b = (const float*)d_in[22];
    const float* ln2g = (const float*)d_in[23], *ln2b = (const float*)d_in[24];
    const float* ln3g = (const float*)d_in[25], *ln3b = (const float*)d_in[26];

    unsigned short* ws = (unsigned short*)d_ws;
    unsigned short* pW1   = ws;               // 12*8*512 = 49152
    unsigned short* pW2   = pW1  + 49152;     // 16384
    unsigned short* pW3   = pW2  + 16384;
    unsigned short* pW11  = pW3  + 16384;
    unsigned short* pW12  = pW11 + 49152;
    unsigned short* pW13  = pW12 + 16384;
    unsigned short* pWin  = pW13 + 16384;     // 4*32*512 = 65536
    unsigned short* pWout = pWin + 65536;     // 16*8*512 = 65536
    float* dh = (float*)(pWout + 65536);      // B*L*H f32

    float* hVout = (float*)d_out;
    float* hEout = hVout + (size_t)B_ * L_ * H_;

    k_pack<<<64, 256, 0, stream>>>(W1w,   pW1,   H_,  IN1);
    k_pack<<<64, 256, 0, stream>>>(W2w,   pW2,   H_,  H_);
    k_pack<<<64, 256, 0, stream>>>(W3w,   pW3,   H_,  H_);
    k_pack<<<64, 256, 0, stream>>>(W11w,  pW11,  H_,  IN1);
    k_pack<<<64, 256, 0, stream>>>(W12w,  pW12,  H_,  H_);
    k_pack<<<64, 256, 0, stream>>>(W13w,  pW13,  H_,  H_);
    k_pack<<<64, 256, 0, stream>>>(Winw,  pWin,  FF_, H_);
    k_pack<<<64, 256, 0, stream>>>(Woutw, pWout, H_,  FF_);

    size_t dynBytes = 2 * (size_t)WBYTES;  // 2 waves per block
    k_node_msg<<<(B_ * L_) / 2, 64, dynBytes, stream>>>(hV, hE, eIdx, maskAtt,
                                                        pW1, W1b, pW2, W2b, pW3, W3b, dh);
    k_node_ffn<<<(B_ * L_) / 32, 64, 0, stream>>>(hV, dh, maskV, pWin, Winb, pWout, Woutb,
                                                  ln1g, ln1b, ln2g, ln2b, hVout);
    k_edge<<<(B_ * L_) / 2, 64, dynBytes, stream>>>(hVout, hE, eIdx,
                                                    pW11, W11b, pW12, W12b, pW13, W13b,
                                                    ln3g, ln3b, hEout);
}
